// LSTM_69346541961901
// MI455X (gfx1250) — matrix operand order repaired and checked by reading, not compiled, not run on hardware
//
#include <hip/hip_runtime.h>
#include <hip/hip_bf16.h>
#include <stdint.h>

// ---------------- types ----------------
typedef __attribute__((ext_vector_type(16))) __bf16 v16bf;
typedef __attribute__((ext_vector_type(8)))  __bf16 v8bf;
typedef __attribute__((ext_vector_type(8)))  float  v8f;
typedef __attribute__((ext_vector_type(4)))  int    v4i;

#define TT 512
#define BB 64
#define HH 1024
#define EE 512
#define KC 128          // K-chunk staged in LDS
#define LSTR (KC + 8)   // padded LDS row stride (u16 elems)

#if __has_builtin(__builtin_amdgcn_global_load_async_to_lds_b128) && \
    __has_builtin(__builtin_amdgcn_s_wait_asynccnt)
#define USE_ASYNC 1
#else
#define USE_ASYNC 0
#endif

// ---------------- helpers ----------------
__device__ __forceinline__ unsigned short f2bf(float f) {
  union { float f; unsigned u; } v; v.f = f;
  unsigned u = v.u;
  return (unsigned short)((u + 0x7fffu + ((u >> 16) & 1u)) >> 16);  // RNE
}
__device__ __forceinline__ float bf2f(unsigned short h) {
  union { unsigned u; float f; } v; v.u = ((unsigned)h) << 16; return v.f;
}
__device__ __forceinline__ float sigm(float x) { return 1.0f / (1.0f + __expf(-x)); }

// Grid-wide barrier for co-resident persistent workgroups.
__device__ __forceinline__ void grid_barrier(unsigned* cnt, unsigned* gen, unsigned nwg) {
  __syncthreads();
  if (threadIdx.x == 0) {
    __threadfence();
    unsigned g = __hip_atomic_load(gen, __ATOMIC_ACQUIRE, __HIP_MEMORY_SCOPE_AGENT);
    if (atomicAdd(cnt, 1u) == nwg - 1u) {
      __hip_atomic_store(cnt, 0u, __ATOMIC_RELAXED, __HIP_MEMORY_SCOPE_AGENT);
      __hip_atomic_fetch_add(gen, 1u, __ATOMIC_RELEASE, __HIP_MEMORY_SCOPE_AGENT);
    } else {
      while (__hip_atomic_load(gen, __ATOMIC_ACQUIRE, __HIP_MEMORY_SCOPE_AGENT) == g)
        __builtin_amdgcn_s_sleep(1);
    }
    __threadfence();
  }
  __syncthreads();
}

// ---------------- parallel embedding gather -> bf16 ----------------
// xg[(t*B+b)*E + k] = bf16(emb[tokens[t,b]*E + k]); fully parallel over T*B*E.
__global__ void emb_gather(const int* __restrict__ tokens, const float* __restrict__ emb,
                           unsigned short* __restrict__ xg) {
  const int nvec = (TT * BB * EE) / 4;
  int i = blockIdx.x * blockDim.x + threadIdx.x;
  if (i >= nvec) return;
  const int row  = i / (EE / 4);
  const int col4 = i % (EE / 4);
  const int tok  = tokens[row];
  float4 v = ((const float4*)(emb + (size_t)tok * EE))[col4];
  unsigned long long p = (unsigned long long)f2bf(v.x) |
                         ((unsigned long long)f2bf(v.y) << 16) |
                         ((unsigned long long)f2bf(v.z) << 32) |
                         ((unsigned long long)f2bf(v.w) << 48);
  *(unsigned long long*)(xg + (size_t)row * EE + col4 * 4) = p;
}

// ---------------- prep: bf16 weight concat, bias sums, state/barrier init ----------------
__global__ void lstm_prep(const float* __restrict__ Wih0, const float* __restrict__ Whh0,
                          const float* __restrict__ bih0, const float* __restrict__ bhh0,
                          const float* __restrict__ Wih1, const float* __restrict__ Whh1,
                          const float* __restrict__ bih1, const float* __restrict__ bhh1,
                          unsigned short* __restrict__ Wcat0, unsigned short* __restrict__ Wcat1,
                          float* __restrict__ bias0, float* __restrict__ bias1,
                          unsigned short* __restrict__ hpA, unsigned short* __restrict__ hpB,
                          float* __restrict__ cbuf, unsigned* __restrict__ bar, long total) {
  const long n0 = 4096L * (EE + HH);
  const long n1 = 4096L * (HH + HH);
  long stride = (long)gridDim.x * blockDim.x;
  for (long i = (long)blockIdx.x * blockDim.x + threadIdx.x; i < total; i += stride) {
    long x = i;
    if (x < n0) {
      long n = x / (EE + HH), k = x % (EE + HH);
      float v = (k < EE) ? Wih0[n * EE + k] : Whh0[n * HH + (k - EE)];
      Wcat0[x] = f2bf(v);
    } else if ((x -= n0) < n1) {
      long n = x / (2 * HH), k = x % (2 * HH);
      float v = (k < HH) ? Wih1[n * HH + k] : Whh1[n * HH + (k - HH)];
      Wcat1[x] = f2bf(v);
    } else if ((x -= n1) < 4096) {
      bias0[x] = bih0[x] + bhh0[x];
    } else if ((x -= 4096) < 4096) {
      bias1[x] = bih1[x] + bhh1[x];
    } else if ((x -= 4096) < BB * HH) {
      cbuf[x] = 0.0f;
    } else if ((x -= BB * HH) < BB * HH) {
      hpA[x] = 0; hpB[x] = 0;
    } else if ((x -= BB * HH) < 64) {
      bar[x] = 0u;
    }
  }
}

// ---------------- A-chunk staging (async-to-LDS when available) ----------------
template <int LAYER>
__device__ __forceinline__ void stage_chunk(unsigned short* __restrict__ dst0,    // &As[buf][0]
                                            const unsigned short* __restrict__ xg_t,
                                            const unsigned short* __restrict__ hist_t,
                                            const unsigned short* __restrict__ hp_r,
                                            int kc) {
  const int b  = threadIdx.x >> 1;
  const int k0 = (threadIdx.x & 1) * 64;
  const unsigned short* src;
  if (LAYER == 0) src = (kc < EE) ? (xg_t + b * EE + kc + k0)
                                  : (hp_r + b * HH + (kc - EE) + k0);
  else            src = (kc < HH) ? (hist_t + b * HH + kc + k0)
                                  : (hp_r + b * HH + (kc - HH) + k0);
  unsigned short* dst = dst0 + b * LSTR + k0;
#if USE_ASYNC
  const __attribute__((address_space(1))) char* g =
      (const __attribute__((address_space(1))) char*)src;
  __attribute__((address_space(3))) char* l =
      (__attribute__((address_space(3))) char*)dst;
#pragma unroll
  for (int i = 0; i < 8; ++i)
    __builtin_amdgcn_global_load_async_to_lds_b128(
        (__attribute__((address_space(1))) v4i*)(g + i * 16),
        (__attribute__((address_space(3))) v4i*)(l + i * 16), 0, 0);
#else
  const uint4* s4 = (const uint4*)src;
  uint4* d4 = (uint4*)dst;
#pragma unroll
  for (int i = 0; i < 8; ++i) d4[i] = s4[i];
#endif
}

__device__ __forceinline__ void wait_stage(bool more_inflight) {
#if USE_ASYNC
  if (more_inflight) __builtin_amdgcn_s_wait_asynccnt(8);
  else               __builtin_amdgcn_s_wait_asynccnt(0);
#else
  (void)more_inflight;
#endif
}

// ---------------- persistent recurrent layer ----------------
// 64 WGs x 128 threads. WG w owns h columns [16w, 16w+16). Each of the 4 waves
// owns a 16-row (batch) block and accumulates 4 gate tiles (16x16 f32 each).
template <int LAYER>
__global__ void __launch_bounds__(128)
lstm_rec(const unsigned short* __restrict__ xg,
         const unsigned short* __restrict__ Wcat, const float* __restrict__ bias,
         unsigned short* __restrict__ hpA, unsigned short* __restrict__ hpB,
         float* __restrict__ cbuf, unsigned short* __restrict__ hist,
         unsigned* __restrict__ bar) {
  constexpr int K = (LAYER == 0) ? (EE + HH) : (2 * HH);

  __shared__ __align__(16) unsigned short As[2][BB * LSTR];

  const int j0    = blockIdx.x * 16;
  const int wave  = threadIdx.x >> 5;
  const int lane  = threadIdx.x & 31;
  const int lhalf = lane >> 4;
  const int lmod  = lane & 15;
  unsigned* cnt = bar;
  unsigned* gen = bar + 16;

  for (int i = threadIdx.x; i < BB * 16; i += 128) {
    int b = i >> 4, c = i & 15;
    cbuf[b * HH + j0 + c] = 0.0f;
    hpA[b * HH + j0 + c]  = 0;
  }
  grid_barrier(cnt, gen, gridDim.x);

  const float bi0 = bias[0 * HH + j0 + lmod];
  const float bf0 = bias[1 * HH + j0 + lmod];
  const float bg0 = bias[2 * HH + j0 + lmod];
  const float bo0 = bias[3 * HH + j0 + lmod];

  for (int t = 0; t < TT; ++t) {
    const unsigned short* hp_r = (t & 1) ? hpB : hpA;   // h_{t-1}
    unsigned short*       hp_w = (t & 1) ? hpA : hpB;   // h_t
    const unsigned short* xg_t   = xg + (size_t)t * BB * EE;
    const unsigned short* hist_t = hist + (size_t)t * BB * HH;

    v8f acc[4];
#pragma unroll
    for (int g = 0; g < 4; ++g)
#pragma unroll
      for (int r = 0; r < 8; ++r) acc[g][r] = 0.0f;

    // prologue: kick chunk 0 (never reads hp_r: kc=0 < EE/HH)
    stage_chunk<LAYER>(&As[0][0], xg_t, hist_t, hp_r, 0);

    int buf = 0;
    for (int kc = 0; kc < K; kc += KC) {
      const bool has_next = (kc + KC) < K;
      if (has_next) stage_chunk<LAYER>(&As[buf ^ 1][0], xg_t, hist_t, hp_r, kc + KC);
      wait_stage(has_next);
      __syncthreads();                       // chunk `buf` fully in LDS, WG-wide

      const unsigned short* arow = &As[buf][(wave * 16 + lmod) * LSTR];
#pragma unroll
      for (int ks = 0; ks < KC; ks += 32) {
        v8bf a0 = *(const v8bf*)(arow + ks + lhalf * 16);
        v8bf a1 = *(const v8bf*)(arow + ks + lhalf * 16 + 8);
        v16bf a = __builtin_shufflevector(a0, a1, 0, 1, 2, 3, 4, 5, 6, 7,
                                          8, 9, 10, 11, 12, 13, 14, 15);
#pragma unroll
        for (int g = 0; g < 4; ++g) {
          const unsigned short* bp =
              Wcat + (size_t)(g * HH + j0 + lmod) * K + kc + ks + lhalf * 16;
          v16bf bfrag = *(const v16bf*)bp;
          acc[g] = __builtin_amdgcn_wmma_f32_16x16x32_bf16(
              false, a, false, bfrag, (short)0, acc[g], false, false);
        }
      }
      __syncthreads();                       // all waves done reading As[buf]
      buf ^= 1;
    }

    // ---- register-local LSTM elementwise (C layout: lane=col, VGPR=row) ----
#pragma unroll
    for (int r = 0; r < 8; ++r) {
      const int b   = wave * 16 + lhalf * 8 + r;
      const int idx = b * HH + j0 + lmod;
      float iv = sigm(acc[0][r] + bi0);
      float fv = sigm(acc[1][r] + bf0);
      float gv = tanhf(acc[2][r] + bg0);
      float ov = sigm(acc[3][r] + bo0);
      float cn = fv * cbuf[idx] + iv * gv;
      cbuf[idx] = cn;
      unsigned short hb = f2bf(ov * tanhf(cn));
      hp_w[idx] = hb;
      if (LAYER == 0) hist[(size_t)t * BB * HH + idx] = hb;
    }
    grid_barrier(cnt, gen, gridDim.x);
  }
}

// ---------------- final dense + 10*sigmoid ----------------
__global__ void lstm_dense(const unsigned short* __restrict__ hlast,
                           const float* __restrict__ Wd, const float* __restrict__ bd,
                           float* __restrict__ out) {
  int b = threadIdx.x;
  if (b >= BB) return;
  float s = 0.0f;
  for (int k = 0; k < HH; ++k) s += bf2f(hlast[b * HH + k]) * Wd[k];
  s += bd[0];
  out[b] = 10.0f / (1.0f + __expf(-s));
}

// ---------------- launch ----------------
extern "C" void kernel_launch(void* const* d_in, const int* in_sizes, int n_in,
                              void* d_out, int out_size, void* d_ws, size_t ws_size,
                              hipStream_t stream) {
  const int*   tokens = (const int*)d_in[0];
  const float* emb    = (const float*)d_in[1];
  const float* Wih0   = (const float*)d_in[2];
  const float* Whh0   = (const float*)d_in[3];
  const float* bih0   = (const float*)d_in[4];
  const float* bhh0   = (const float*)d_in[5];
  const float* Wih1   = (const float*)d_in[6];
  const float* Whh1   = (const float*)d_in[7];
  const float* bih1   = (const float*)d_in[8];
  const float* bhh1   = (const float*)d_in[9];
  const float* Wd     = (const float*)d_in[10];
  const float* bd     = (const float*)d_in[11];

  char* ws = (char*)d_ws;
  size_t o = 0;
  auto take = [&](size_t bytes) { size_t r = o; o = (o + bytes + 255) & ~(size_t)255; return r; };

  unsigned*       bar   = (unsigned*)(ws + take(256));
  unsigned short* Wcat0 = (unsigned short*)(ws + take(4096UL * (EE + HH) * 2));
  unsigned short* Wcat1 = (unsigned short*)(ws + take(4096UL * (2 * HH) * 2));
  float*          bias0 = (float*)(ws + take(4096UL * 4));
  float*          bias1 = (float*)(ws + take(4096UL * 4));
  unsigned short* hpA   = (unsigned short*)(ws + take((size_t)BB * HH * 2));
  unsigned short* hpB   = (unsigned short*)(ws + take((size_t)BB * HH * 2));
  float*          cbuf  = (float*)(ws + take((size_t)BB * HH * 4));
  unsigned short* hist  = (unsigned short*)(ws + take((size_t)TT * BB * HH * 2));
  unsigned short* xg    = (unsigned short*)(ws + take((size_t)TT * BB * EE * 2));

  const long total = 4096L * (EE + HH) + 4096L * (2 * HH) + 4096 + 4096 +
                     (long)BB * HH + (long)BB * HH + 64;
  lstm_prep<<<2048, 256, 0, stream>>>(Wih0, Whh0, bih0, bhh0, Wih1, Whh1, bih1, bhh1,
                                      Wcat0, Wcat1, bias0, bias1, hpA, hpB, cbuf, bar, total);

  const int gvec = (TT * BB * EE / 4 + 255) / 256;
  emb_gather<<<gvec, 256, 0, stream>>>(tokens, emb, xg);

  lstm_rec<0><<<64, 128, 0, stream>>>(xg, Wcat0, bias0, hpA, hpB, cbuf, hist, bar);
  lstm_rec<1><<<64, 128, 0, stream>>>(xg, Wcat1, bias1, hpA, hpB, cbuf, hist, bar);

  // T=512 (even): final h of each layer lands in hpA
  lstm_dense<<<1, 64, 0, stream>>>(hpA, Wd, bd, (float*)d_out);
}
